// StosaKT_86208583565882
// MI455X (gfx1250) — compile-verified, matrix operand order checked by reading, not run on hardware
//
#include <hip/hip_runtime.h>
#include <hip/hip_bf16.h>

// ---------------- constants ----------------
#define Bsz   16
#define Lseq  512
#define Hdim  512
#define NH    8
#define DH    64
#define NQc   10000
#define MROWS (Bsz * Lseq)          // 8192
#define MASKV (-4294967295.0f)      // -2^32 + 1

typedef __attribute__((ext_vector_type(16))) _Float16 v16h;
typedef __attribute__((ext_vector_type(8)))  float    v8f;
typedef unsigned int u32;
typedef u32 __attribute__((ext_vector_type(4))) v4u;
typedef int __attribute__((ext_vector_type(4))) v4i;
typedef int __attribute__((ext_vector_type(8))) v8i;

// ---------------- CDNA5 primitives ----------------
__device__ __forceinline__ v8f wmma32(v16h a, v16h b, v8f c) {
  // D = A x B + C ; args: (neg_a, A, neg_b, B, c_mod, C, reuse_a, reuse_b)
  return __builtin_amdgcn_wmma_f32_16x16x32_f16(false, a, false, b, (short)0, c, false, false);
}

// async global -> LDS copy, 16B per lane (GV mode, ASYNCcnt-tracked)
__device__ __forceinline__ void async_copy16(unsigned lds_off, const float* g) {
  asm volatile("global_load_async_to_lds_b128 %0, %1, off"
               :: "v"(lds_off), "v"(g) : "memory");
}
__device__ __forceinline__ void wait_async4() {
  asm volatile("s_wait_asynccnt 0x4" ::: "memory");
}
__device__ __forceinline__ void wait_async0() {
  asm volatile("s_wait_asynccnt 0x0" ::: "memory");
}

// TDM: 2D tile load global->LDS. tile_w elems/row, tile_h rows, row stride (elems).
// D# layout per ISA 8.3/8.4: group0 {count=1, lds_addr, global_addr, type=2},
// group1 {wg_mask=0, data_size=2(4B), tensor_dim0/1, tile_dim0/1, dim0_stride}.
// Descriptor words are wave-uniform; readfirstlane forces SGPR class.
// This toolchain exposes the 6-arg builtin: (v4u, v8i, v4i, v4i, v8i, i32 cpol).
__device__ __forceinline__ void tdm_load_tile(unsigned lds_addr, const float* gp,
                                              u32 tile_w, u32 tile_h, u32 stride_elems,
                                              u32 td0, u32 td1) {
  unsigned long long ga = (unsigned long long)gp;
  u32 lo = (u32)__builtin_amdgcn_readfirstlane((int)(u32)ga);
  u32 hi = (u32)__builtin_amdgcn_readfirstlane((int)(u32)(ga >> 32));
  u32 ld = (u32)__builtin_amdgcn_readfirstlane((int)lds_addr);
  v4u g0 = { 1u, ld, lo, (hi & 0x1FFFFFFu) | (2u << 30) };
  v8i g1 = { (int)(2u << 16),                                            // data_size=4B
             (int)((td0 & 0xFFFFu) << 16),                               // tensor_dim0 lo
             (int)(((td0 >> 16) & 0xFFFFu) | ((td1 & 0xFFFFu) << 16)),   // td0 hi | td1 lo
             (int)(((td1 >> 16) & 0xFFFFu) | (tile_w << 16)),            // td1 hi | tile_dim0
             (int)tile_h,                                                // tile_dim1 (tile_dim2=0)
             (int)stride_elems,                                          // dim0_stride lo
             0, 0 };
  v4i gz4 = { 0, 0, 0, 0 };
  v8i gz8 = { 0, 0, 0, 0, 0, 0, 0, 0 };
  __builtin_amdgcn_tensor_load_to_lds(g0, g1, gz4, gz4, gz8, 0);
}

// Fragment builders. ISA 7.12.2 16-bit A layout: lane = row (0..15), lanes 16-31 K+=8,
// element j -> K = (j/8)*16 + (lane/16)*8 + (j%8). B mirrors with lane = col.
__device__ __forceinline__ v16h ldfrag(const _Float16* s, int r0, int k0, int ld) {
  int lane = threadIdx.x & 31;
  const _Float16* p = s + (long)(r0 + (lane & 15)) * ld + k0 + ((lane >> 4) << 3);
  v16h f;
#pragma unroll
  for (int j = 0; j < 8; ++j) { f[j] = p[j]; f[j + 8] = p[j + 16]; }
  return f;
}
// from f32 LDS tile (row-major [row][32]); optional squaring (for P^2 @ Vc)
__device__ __forceinline__ v16h ldfrag32(const float* s, int r0, bool sq) {
  int lane = threadIdx.x & 31;
  const float* p = s + (long)(r0 + (lane & 15)) * 32 + ((lane >> 4) << 3);
  v16h f;
#pragma unroll
  for (int j = 0; j < 8; ++j) {
    float a = p[j], b = p[j + 16];
    if (sq) { a *= a; b *= b; }
    f[j] = (_Float16)a; f[j + 8] = (_Float16)b;
  }
  return f;
}

__device__ __forceinline__ float eluf(float x) { return x > 0.f ? x : (expf(x) - 1.f); }

__device__ __forceinline__ float blk_reduce(float v, float* sm) {
  int t = threadIdx.x;
  sm[t] = v; __syncthreads();
  for (int s = blockDim.x >> 1; s > 0; s >>= 1) {
    if (t < s) sm[t] += sm[t + s];
    __syncthreads();
  }
  float r = sm[0]; __syncthreads();
  return r;
}

// ---------------- GEMM: C[M,N] = epi(A[M,K] @ W[K,N] + bias) (+resid) ----------------
// EPI: 0=none, 1=elu(x)+1, 2=exact gelu. Block tile 128x64x32, 256 threads (8 waves).
// A tiles stream via TDM (one 16x32 slice per wave, double buffered, TENSORcnt);
// W tiles via VGPR prefetch -> transposed f16 LDS.
template<int EPI>
__global__ __launch_bounds__(256) void k_gemm(const float* __restrict__ A,
                                              const float* __restrict__ W,
                                              const float* __restrict__ bias,
                                              const float* __restrict__ resid,
                                              float* __restrict__ C,
                                              int M, int N, int K) {
  __shared__ float    As[2][128 * 32];   // [m][k] raw f32, TDM-filled
  __shared__ _Float16 Bs[2][64 * 32];    // transposed [n][k] f16
  const int t = threadIdx.x, lane = t & 31, w = t >> 5;
  const int m0 = blockIdx.x * 128, n0 = blockIdx.y * 64;
  const int wr = (w >> 1) * 32, wc = (w & 1) * 32;
  const int KT = K >> 5;

  float breg[8];
  auto issueA = [&](int buf, int kt) {               // each wave DMAs its 16-row slice
    const int k0 = kt << 5;
    tdm_load_tile((unsigned)(size_t)&As[buf][w * 512],
                  &A[(long)(m0 + w * 16) * K + k0],
                  32u, 16u, (u32)K, (u32)K, (u32)M);
  };
  auto loadB = [&](int kt) {
    const int k0 = kt << 5;
#pragma unroll
    for (int ii = 0; ii < 8; ++ii) {
      int i = t + ii * 256;
      int r = i >> 6, c = i & 63;                    // r: k-row, c: n-col
      breg[ii] = W[(long)(k0 + r) * N + n0 + c];
    }
  };
  auto storeB = [&](int buf) {
#pragma unroll
    for (int ii = 0; ii < 8; ++ii) {
      int i = t + ii * 256;
      int r = i >> 6, c = i & 63;
      Bs[buf][c * 32 + r] = (_Float16)breg[ii];
    }
  };

  issueA(0, 0);
  loadB(0);
  v8f acc[2][2] = {};
  for (int kt = 0; kt < KT; ++kt) {
    const int cur = kt & 1;
    if (kt + 1 < KT) issueA(cur ^ 1, kt + 1);
    storeB(cur);
    if (kt + 1 < KT) { loadB(kt + 1); __builtin_amdgcn_s_wait_tensorcnt(1); }
    else             { __builtin_amdgcn_s_wait_tensorcnt(0); }
    __syncthreads();
    v16h a0 = ldfrag32(As[cur], wr, false), a1 = ldfrag32(As[cur], wr + 16, false);
    v16h b0 = ldfrag(Bs[cur], wc, 0, 32),   b1 = ldfrag(Bs[cur], wc + 16, 0, 32);
    acc[0][0] = wmma32(a0, b0, acc[0][0]);
    acc[0][1] = wmma32(a0, b1, acc[0][1]);
    acc[1][0] = wmma32(a1, b0, acc[1][0]);
    acc[1][1] = wmma32(a1, b1, acc[1][1]);
    __syncthreads();
  }
  const int colb = lane & 15, rb = (lane >> 4) << 3;
#pragma unroll
  for (int ti = 0; ti < 2; ++ti)
#pragma unroll
    for (int tj = 0; tj < 2; ++tj) {
      int col = n0 + wc + tj * 16 + colb;
      float bv = bias ? bias[col] : 0.f;
#pragma unroll
      for (int r = 0; r < 8; ++r) {
        int row = m0 + wr + ti * 16 + rb + r;
        float v = acc[ti][tj][r] + bv;
        if (resid) v += resid[(long)row * N + col];
        if (EPI == 1) v = eluf(v) + 1.f;
        if (EPI == 2) v = 0.5f * v * (1.f + erff(v * 0.70710678f));
        C[(long)row * N + col] = v;
      }
    }
}

// ---------------- embeddings + LN + ELU ----------------
// grid (8192 rows, 4 streams): 0 mean_seq(ids), 1 mean_resp(qa), 2 cov_seq(ids,+1), 3 cov_resp(ids,+1 faithful bug)
__global__ __launch_bounds__(256) void k_embed(const int* cs, const int* rs, const int* shc, const int* shr,
    const float* im, const float* irm, const float* ic, const float* irc,
    const float* pm, const float* prm, const float* pc, const float* prc,
    const float* g, const float* bb,
    float* mean_seq, float* mean_resp, float* cov_seq, float* cov_resp,
    float* mean_h, float* cov_h) {
  __shared__ float sm[256];
  const int row = blockIdx.x, st = blockIdx.y, t = threadIdx.x;
  const int b = row >> 9, l = row & 511;
  const int ids = l ? shc[b * 511 + l - 1] : cs[b * 511];
  const int tgt = l ? shr[b * 511 + l - 1] : rs[b * 511];
  const float* tab; const float* pos; int idx; int plus1 = 0;
  float* outb; float* extra = nullptr;
  switch (st) {
    case 0:  tab = im;  pos = pm;  idx = ids;                 outb = mean_seq; extra = mean_h; break;
    case 1:  tab = irm; pos = prm; idx = ids + NQc * tgt;     outb = mean_resp;               break;
    case 2:  tab = ic;  pos = pc;  idx = ids; plus1 = 1;      outb = cov_seq;  extra = cov_h; break;
    default: tab = irc; pos = prc; idx = ids; plus1 = 1;      outb = cov_resp;                break;
  }
  float x0 = tab[(long)idx * Hdim + t]       + pos[(long)l * Hdim + t];
  float x1 = tab[(long)idx * Hdim + t + 256] + pos[(long)l * Hdim + t + 256];
  float u = blk_reduce(x0 + x1, sm) * (1.f / Hdim);
  float v = blk_reduce((x0 - u) * (x0 - u) + (x1 - u) * (x1 - u), sm) * (1.f / Hdim);
  float inv = rsqrtf(v + 1e-12f);
  float y0 = eluf((x0 - u) * inv * g[t]       + bb[t])       + (plus1 ? 1.f : 0.f);
  float y1 = eluf((x1 - u) * inv * g[t + 256] + bb[t + 256]) + (plus1 ? 1.f : 0.f);
  long o = (long)row * Hdim;
  outb[o + t] = y0; outb[o + t + 256] = y1;
  if (extra) { extra[o + t] = y0; extra[o + t + 256] = y1; }
}

// ---------------- row stats: qstat = sum qm^2 + sum qc ; kstat likewise ----------------
__global__ void k_rowstats(const float* qm, const float* qc, const float* km, const float* kc,
                           float* qstat, float* kstat) {
  int i = blockIdx.x * blockDim.x + threadIdx.x;       // i = bh*L + l
  if (i >= Bsz * NH * Lseq) return;
  int bh = i / Lseq, l = i % Lseq, b = bh / NH, h = bh % NH;
  long base = ((long)b * Lseq + l) * Hdim + h * DH;
  float sq = 0.f, sk = 0.f;
#pragma unroll 8
  for (int d = 0; d < DH; ++d) {
    float mq = qm[base + d], mk = km[base + d];
    sq += mq * mq + qc[base + d];
    sk += mk * mk + kc[base + d];
  }
  qstat[i] = sq; kstat[i] = sk;
}

// ---------------- scores: -(qstat+kstat-2*(qm.km + sqrt(qc).sqrt(kc)))*scale + mask ----------------
// grid (L/64, L/64, B*NH); 64q x 64k tile per block, K-dim = DH = 64.
__global__ __launch_bounds__(256) void k_scores(const float* qm, const float* qc,
                                                const float* km, const float* kc,
                                                const float* qstat, const float* kstat,
                                                float* __restrict__ out) {
  __shared__ _Float16 Qm[64 * 64], Qs[64 * 64], Km[64 * 64], Ks[64 * 64];
  const int t = threadIdx.x, lane = t & 31, w = t >> 5;
  const int q0 = blockIdx.x * 64, k0 = blockIdx.y * 64;
  const int bh = blockIdx.z, b = bh >> 3, h = bh & 7;
  const long qb = ((long)b * Lseq) * Hdim + h * DH;
#pragma unroll 4
  for (int i = t; i < 64 * 64; i += 256) {
    int r = i >> 6, d = i & 63;
    Qm[i] = (_Float16)qm[qb + (long)(q0 + r) * Hdim + d];
    Qs[i] = (_Float16)sqrtf(fmaxf(qc[qb + (long)(q0 + r) * Hdim + d], 1e-24f));
    Km[i] = (_Float16)km[qb + (long)(k0 + r) * Hdim + d];
    Ks[i] = (_Float16)sqrtf(fmaxf(kc[qb + (long)(k0 + r) * Hdim + d], 1e-24f));
  }
  __syncthreads();
  const int tr = w >> 1;                       // row tile 0..3
  const int tc0 = (w & 1) * 2;                 // col tiles {tc0, tc0+1}
  const int colb = lane & 15, rb = (lane >> 4) << 3;
#pragma unroll
  for (int tc = 0; tc < 2; ++tc) {
    int tcg = tc0 + tc;
    v8f acc = {};
#pragma unroll
    for (int kk = 0; kk < 64; kk += 32) {
      acc = wmma32(ldfrag(Qm, tr * 16, kk, 64), ldfrag(Km, tcg * 16, kk, 64), acc);
      acc = wmma32(ldfrag(Qs, tr * 16, kk, 64), ldfrag(Ks, tcg * 16, kk, 64), acc);
    }
#pragma unroll
    for (int r = 0; r < 8; ++r) {
      int q = q0 + tr * 16 + rb + r;
      int k = k0 + tcg * 16 + colb;
      float w2 = qstat[bh * Lseq + q] + kstat[bh * Lseq + k] - 2.f * acc[r];
      float s = -w2 * 0.125f + ((k < q) ? 0.f : MASKV);   // scale = 1/sqrt(64)
      out[((long)bh * Lseq + q) * Lseq + k] = s;
    }
  }
}

// ---------------- softmax over last dim, in place. grid = B*NH*L, 128 threads ----------------
__global__ __launch_bounds__(128) void k_softmax(float* __restrict__ s) {
  __shared__ float sm[128];
  const int t = threadIdx.x;
  float* p = s + (long)blockIdx.x * Lseq;
  float x0 = p[t], x1 = p[t + 128], x2 = p[t + 256], x3 = p[t + 384];
  float m = fmaxf(fmaxf(x0, x1), fmaxf(x2, x3));
  sm[t] = m; __syncthreads();
  for (int st = 64; st > 0; st >>= 1) { if (t < st) sm[t] = fmaxf(sm[t], sm[t + st]); __syncthreads(); }
  m = sm[0]; __syncthreads();
  x0 = expf(x0 - m); x1 = expf(x1 - m); x2 = expf(x2 - m); x3 = expf(x3 - m);
  float sum = blk_reduce(x0 + x1 + x2 + x3, sm);
  float inv = 1.f / sum;
  p[t] = x0 * inv; p[t + 128] = x1 * inv; p[t + 256] = x2 * inv; p[t + 384] = x3 * inv;
}

// ---------------- context: O[b,:,h*64+d] = P(^2) @ V[b,:,h*64+d] ----------------
// grid (L/128, 1, B*NH); block tile 128q x 64d, K = 512 keys. P tiles async->LDS f32,
// squaring (cov path) applied during f32->f16 fragment conversion.
__global__ __launch_bounds__(256) void k_attn_ctx(const float* __restrict__ P,
                                                  const float* __restrict__ V,
                                                  float* __restrict__ O, int squared) {
  __shared__ float    As[2][128 * 32];   // [q][k] raw probs
  __shared__ _Float16 Bs[2][64 * 32];    // transposed [d][k]
  const int t = threadIdx.x, lane = t & 31, w = t >> 5;
  const int bh = blockIdx.z, b = bh >> 3, h = bh & 7;
  const int m0 = blockIdx.x * 128;
  const float* Pb = P + (long)bh * Lseq * Lseq;
  const float* Vb = V + (long)b * Lseq * Hdim + h * DH;
  float* Ob = O + (long)b * Lseq * Hdim + h * DH;
  const int wr = (w >> 1) * 32, wc = (w & 1) * 32;
  const bool sq = squared != 0;
  const int KT = Lseq >> 5;              // 16

  float breg[8];
  auto issueA = [&](int buf, int kt) {
    const int k0 = kt << 5;
#pragma unroll
    for (int ii = 0; ii < 4; ++ii) {
      int i = (t + ii * 256) << 2;
      int r = i >> 5, c = i & 31;
      async_copy16((unsigned)(size_t)&As[buf][i], &Pb[(long)(m0 + r) * Lseq + k0 + c]);
    }
  };
  auto loadB = [&](int kt) {
    const int k0 = kt << 5;
#pragma unroll
    for (int ii = 0; ii < 8; ++ii) {
      int i = t + ii * 256;
      int r = i >> 6, c = i & 63;        // r: key row, c: d col
      breg[ii] = Vb[(long)(k0 + r) * Hdim + c];
    }
  };
  auto storeB = [&](int buf) {
#pragma unroll
    for (int ii = 0; ii < 8; ++ii) {
      int i = t + ii * 256;
      int r = i >> 6, c = i & 63;
      Bs[buf][c * 32 + r] = (_Float16)breg[ii];
    }
  };

  issueA(0, 0);
  loadB(0);
  v8f acc[2][2] = {};
  for (int kt = 0; kt < KT; ++kt) {
    const int cur = kt & 1;
    if (kt + 1 < KT) issueA(cur ^ 1, kt + 1);
    storeB(cur);
    if (kt + 1 < KT) { loadB(kt + 1); wait_async4(); } else { wait_async0(); }
    __syncthreads();
    v16h a0 = ldfrag32(As[cur], wr, sq), a1 = ldfrag32(As[cur], wr + 16, sq);
    v16h b0 = ldfrag(Bs[cur], wc, 0, 32), b1 = ldfrag(Bs[cur], wc + 16, 0, 32);
    acc[0][0] = wmma32(a0, b0, acc[0][0]);
    acc[0][1] = wmma32(a0, b1, acc[0][1]);
    acc[1][0] = wmma32(a1, b0, acc[1][0]);
    acc[1][1] = wmma32(a1, b1, acc[1][1]);
    __syncthreads();
  }
  const int colb = lane & 15, rb = (lane >> 4) << 3;
#pragma unroll
  for (int ti = 0; ti < 2; ++ti)
#pragma unroll
    for (int tj = 0; tj < 2; ++tj)
#pragma unroll
      for (int r = 0; r < 8; ++r) {
        int row = m0 + wr + ti * 16 + rb + r;
        int col = wc + tj * 16 + colb;
        Ob[(long)row * Hdim + col] = acc[ti][tj][r];
      }
}

// ---------------- layernorm in place, grid 8192 x 256 ----------------
__global__ __launch_bounds__(256) void k_layernorm(float* __restrict__ x,
                                                   const float* g, const float* bb) {
  __shared__ float sm[256];
  const int t = threadIdx.x;
  float* p = x + (long)blockIdx.x * Hdim;
  float x0 = p[t], x1 = p[t + 256];
  float u = blk_reduce(x0 + x1, sm) * (1.f / Hdim);
  float v = blk_reduce((x0 - u) * (x0 - u) + (x1 - u) * (x1 - u), sm) * (1.f / Hdim);
  float inv = rsqrtf(v + 1e-12f);
  p[t]       = (x0 - u) * inv * g[t]       + bb[t];
  p[t + 256] = (x1 - u) * inv * g[t + 256] + bb[t + 256];
}

// ---------------- final wasserstein_elem + sigmoid ----------------
__global__ __launch_bounds__(256) void k_final(const float* mh, const float* ch,
                                               const float* ms, const float* cs,
                                               const float* ow, const float* ob,
                                               float* __restrict__ out) {
  __shared__ float sm[256];
  const int row = blockIdx.x, t = threadIdx.x;
  const int b = row >> 9, l = row & 511;
  long o = (long)row * Hdim;
  float acc = 0.f;
#pragma unroll
  for (int it = 0; it < 2; ++it) {
    int d = t + it * 256;
    float dm = mh[o + d] - ms[o + d];
    float s1 = sqrtf(fmaxf(ch[o + d], 1e-24f));
    float s2 = sqrtf(fmaxf(cs[o + d], 1e-24f));
    acc += dm * dm + (s1 - s2) * (s1 - s2);
  }
  float tot = blk_reduce(acc, sm);
  if (t == 0 && l > 0) {
    float logit = tot * ow[0] + ob[0];
    out[b * 511 + (l - 1)] = 1.f / (1.f + expf(-logit));
  }
}

// ---------------- host ----------------
extern "C" void kernel_launch(void* const* d_in, const int* in_sizes, int n_in,
                              void* d_out, int out_size, void* d_ws, size_t ws_size,
                              hipStream_t stream) {
  (void)in_sizes; (void)n_in; (void)out_size; (void)ws_size;
  const int* cseqs = (const int*)d_in[1];
  const int* rseqs = (const int*)d_in[2];
  const int* shc   = (const int*)d_in[4];
  const int* shr   = (const int*)d_in[5];
  auto F = [&](int i) { return (const float*)d_in[i]; };

  // workspace layout (floats): 16 activation buffers [B*L,H] + stats + scores
  float* ws = (float*)d_ws;
  const long ACT = (long)MROWS * Hdim;
  float* mean_seq  = ws + 0 * ACT;
  float* cov_seq   = ws + 1 * ACT;
  float* mean_resp = ws + 2 * ACT;
  float* cov_resp  = ws + 3 * ACT;
  float* mean_h    = ws + 4 * ACT;
  float* cov_h     = ws + 5 * ACT;
  float* qm = ws + 6 * ACT,  *km = ws + 7 * ACT,  *vm = ws + 8 * ACT;
  float* qc = ws + 9 * ACT,  *kc = ws + 10 * ACT, *vc = ws + 11 * ACT;
  float* qstat  = ws + 16 * ACT;                         // B*NH*L
  float* kstat  = qstat + Bsz * NH * Lseq;
  float* scores = kstat + Bsz * NH * Lseq;               // B*NH*L*L fp32
  float* ffn_tmp = scores;                               // reused after attention

  k_embed<<<dim3(MROWS, 4), 256, 0, stream>>>(cseqs, rseqs, shc, shr,
      F(6), F(7), F(8), F(9), F(10), F(11), F(12), F(13), F(14), F(15),
      mean_seq, mean_resp, cov_seq, cov_resp, mean_h, cov_h);

  const dim3 gp(MROWS / 128, Hdim / 64);          // 64 x 8
  const dim3 g1(MROWS / 128, 2048 / 64);          // 64 x 32
  const dim3 gs(Lseq / 64, Lseq / 64, Bsz * NH);  // 8 x 8 x 128
  const dim3 gc(Lseq / 128, 1, Bsz * NH);         // 4 x 1 x 128

  for (int ly = 0; ly < 2; ++ly) {
    const int p = 18 + ly * 24;
    auto Wp = [&](int j) { return (const float*)d_in[p + j]; };

    k_gemm<0><<<gp, 256, 0, stream>>>(mean_h,    Wp(0),  Wp(1),  nullptr, qm, MROWS, Hdim, Hdim);
    k_gemm<0><<<gp, 256, 0, stream>>>(mean_h,    Wp(2),  Wp(3),  nullptr, km, MROWS, Hdim, Hdim);
    k_gemm<0><<<gp, 256, 0, stream>>>(mean_resp, Wp(4),  Wp(5),  nullptr, vm, MROWS, Hdim, Hdim);
    k_gemm<1><<<gp, 256, 0, stream>>>(cov_h,     Wp(6),  Wp(7),  nullptr, qc, MROWS, Hdim, Hdim);
    k_gemm<1><<<gp, 256, 0, stream>>>(cov_h,     Wp(8),  Wp(9),  nullptr, kc, MROWS, Hdim, Hdim);
    k_gemm<1><<<gp, 256, 0, stream>>>(cov_resp,  Wp(10), Wp(11), nullptr, vc, MROWS, Hdim, Hdim);

    k_rowstats<<<(Bsz * NH * Lseq) / 256, 256, 0, stream>>>(qm, qc, km, kc, qstat, kstat);
    k_scores<<<gs, 256, 0, stream>>>(qm, qc, km, kc, qstat, kstat, scores);
    k_softmax<<<Bsz * NH * Lseq, 128, 0, stream>>>(scores);

    k_attn_ctx<<<gc, 256, 0, stream>>>(scores, vm, qm, 0);   // ctx_m -> qm buf
    k_attn_ctx<<<gc, 256, 0, stream>>>(scores, vc, qc, 1);   // ctx_c -> qc buf

    // attention output proj + residual + LN  (mean_a -> km buf, cov_a -> kc buf)
    k_gemm<0><<<gp, 256, 0, stream>>>(qm, Wp(12), Wp(13), mean_h, km, MROWS, Hdim, Hdim);
    k_layernorm<<<MROWS, 256, 0, stream>>>(km, Wp(16), Wp(17));
    k_gemm<0><<<gp, 256, 0, stream>>>(qc, Wp(14), Wp(15), cov_h, kc, MROWS, Hdim, Hdim);
    k_layernorm<<<MROWS, 256, 0, stream>>>(kc, Wp(16), Wp(17));

    // FFN mean path
    k_gemm<2><<<g1, 256, 0, stream>>>(km, Wp(18), Wp(19), nullptr, ffn_tmp, MROWS, 2048, Hdim);
    k_gemm<0><<<gp, 256, 0, stream>>>(ffn_tmp, Wp(20), Wp(21), km, mean_h, MROWS, Hdim, 2048);
    k_layernorm<<<MROWS, 256, 0, stream>>>(mean_h, Wp(22), Wp(23));
    // FFN cov path
    k_gemm<2><<<g1, 256, 0, stream>>>(kc, Wp(18), Wp(19), nullptr, ffn_tmp, MROWS, 2048, Hdim);
    k_gemm<0><<<gp, 256, 0, stream>>>(ffn_tmp, Wp(20), Wp(21), kc, cov_h, MROWS, Hdim, 2048);
    k_layernorm<<<MROWS, 256, 0, stream>>>(cov_h, Wp(22), Wp(23));
  }

  k_final<<<MROWS, 256, 0, stream>>>(mean_h, cov_h, mean_seq, cov_seq,
                                     F(16), F(17), (float*)d_out);
}
